// VectorQuantizer_51505247813988
// MI455X (gfx1250) — matrix operand order repaired and checked by reading, not compiled, not run on hardware
//
#include <hip/hip_runtime.h>

// VQ argmin via split-bf16 WMMA on gfx1250.
//   score(n,k) = ||c_k||^2 - 2 * x_n . c_k      (x_sq omitted: argmin-invariant)
// A = (-2*x) split into bf16 hi+lo, B = codebook split into bf16 hi+lo,
// dot ~= ah*bh + al*bh + ah*bl  (3x v_wmma_f32_16x16x32_bf16, f32 accum).

typedef __attribute__((ext_vector_type(16))) __bf16 v16bf;
typedef __attribute__((ext_vector_type(8)))  float  v8f;

#define DDIM   256
#define KCODES 8192
#define NROWS  16384
#define MB     32
#define NB     32

__device__ __forceinline__ unsigned short f2bf(float f) {
    unsigned u = __float_as_uint(f);
    unsigned r = u + 0x7FFFu + ((u >> 16) & 1u);   // round-to-nearest-even
    return (unsigned short)(r >> 16);
}
__device__ __forceinline__ float bf2f(unsigned short h) {
    return __uint_as_float((unsigned)h << 16);
}

union Frag { uint4 q[2]; v16bf v; };

// A-frag (16x32 bf16): lane group g: halves = K [g*8, g*8+8) and [g*8+16, g*8+24)
__device__ __forceinline__ v16bf load_fragA(const unsigned short* row, int d0, int g) {
    Frag u;
    const uint4* p = (const uint4*)(row + d0 + g * 8);
    u.q[0] = p[0];
    u.q[1] = p[2];          // +16 halves = +32B
    return u.v;
}
// B-frag (32x16 bf16): lane group g: halves = K [g*8, g*8+8) and [g*8+16, g*8+24)
__device__ __forceinline__ v16bf load_fragB(const unsigned short* row, int d0, int g) {
    Frag u;
    const uint4* p = (const uint4*)(row + d0 + g * 8);
    u.q[0] = p[0];
    u.q[1] = p[2];
    return u.v;
}

// ---------------------------------------------------------------- c_sq[k]
__global__ __launch_bounds__(256) void vq_csq_kernel(const float* __restrict__ cb,
                                                     float* __restrict__ csq) {
    const int lane = threadIdx.x & 31;
    const int wv   = threadIdx.x >> 5;                 // 8 waves -> 8 codes/block
    const int k    = blockIdx.x * 8 + wv;
    const float* row = cb + (long long)k * DDIM;
    float s = 0.f;
#pragma unroll
    for (int j = 0; j < 8; ++j) { float x = row[lane + j * 32]; s += x * x; }
#pragma unroll
    for (int m = 16; m >= 1; m >>= 1) s += __shfl_xor(s, m, 32);
    if (lane == 0) csq[k] = s;
}

// ---------------------------------------------------------------- argmin GEMM
__global__ __launch_bounds__(128) void vq_argmin_kernel(const float* __restrict__ input,
                                                        const float* __restrict__ codebook,
                                                        const float* __restrict__ csq,
                                                        int* __restrict__ idx_out) {
    __shared__ unsigned short Ahi[MB * DDIM];   // 16 KB
    __shared__ unsigned short Alo[MB * DDIM];   // 16 KB
    __shared__ unsigned short Bhi[NB * DDIM];   // 16 KB (reused for reduction)
    __shared__ unsigned short Blo[NB * DDIM];   // 16 KB (reused for reduction)

    const int tid  = threadIdx.x;
    const int lane = tid & 31;
    const int wv   = tid >> 5;        // 0..3
    const int rt   = wv >> 1;         // row tile 0..1
    const int ct   = wv & 1;          // col tile 0..1
    const int g    = lane >> 4;       // half-wave group
    const int ln16 = lane & 15;

    const long long rowBase = (long long)blockIdx.x * MB;

    // ---- load + split A (32 rows x 256 dims), fused with *(-2) and transpose.
    // input layout (B,C,H,W): off = b*262144 + d*1024 + (n%1024); rows contiguous in n.
    {
        const int r_ = tid & 31;
        const int dg = tid >> 5;                        // 0..3
        const long long n  = rowBase + r_;
        const float* src = input + (n >> 10) * 262144 + (n & 1023);
#pragma unroll 4
        for (int i = 0; i < 64; ++i) {
            const int d = dg + (i << 2);
            float x = -2.0f * src[(long long)d * 1024]; // lanes coalesced over rows
            unsigned short hi = f2bf(x);
            unsigned short lo = f2bf(x - bf2f(hi));
            Ahi[r_ * DDIM + d] = hi;
            Alo[r_ * DDIM + d] = lo;
        }
    }

    float bestS[8];
    int   bestI[8];
#pragma unroll
    for (int v = 0; v < 8; ++v) { bestS[v] = 3.4e38f; bestI[v] = 0; }

    const unsigned short* arow_hi = &Ahi[(rt * 16 + ln16) * DDIM];
    const unsigned short* arow_lo = &Alo[(rt * 16 + ln16) * DDIM];
    const unsigned short* brow_hi = &Bhi[(ct * 16 + ln16) * DDIM];
    const unsigned short* brow_lo = &Blo[(ct * 16 + ln16) * DDIM];

    for (int blk = 0; blk < KCODES / NB; ++blk) {
        const int code0 = blk * NB;
        __syncthreads();
        // ---- stage 32 codes x 256 f32 -> bf16 hi/lo in LDS (codebook is L2-resident)
        {
            const int c_    = tid >> 2;                 // 0..31
            const int dbase = (tid & 3) << 6;           // 0,64,128,192
            const float* src = codebook + (long long)(code0 + c_) * DDIM + dbase;
#pragma unroll
            for (int j = 0; j < 16; ++j) {
                float4 f = *(const float4*)(src + j * 4);
                float vals[4] = { f.x, f.y, f.z, f.w };
#pragma unroll
                for (int q = 0; q < 4; ++q) {
                    const int d = dbase + j * 4 + q;
                    unsigned short hi = f2bf(vals[q]);
                    unsigned short lo = f2bf(vals[q] - bf2f(hi));
                    Bhi[c_ * DDIM + d] = hi;
                    Blo[c_ * DDIM + d] = lo;
                }
            }
            if (blk + 1 < KCODES / NB)                  // global_prefetch_b8 next tile
                __builtin_prefetch(codebook + (long long)(code0 + NB + c_) * DDIM + dbase, 0, 1);
        }
        __syncthreads();

        // ---- 16x16 tile per wave, K=256 in 8 chunks of 32, 3 independent accums
        v8f acc0 = {}; v8f acc1 = {}; v8f acc2 = {};
#pragma unroll
        for (int kc = 0; kc < 8; ++kc) {
            const int d0 = kc * 32;
            v16bf ah = load_fragA(arow_hi, d0, g);
            v16bf al = load_fragA(arow_lo, d0, g);
            v16bf bh = load_fragB(brow_hi, d0, g);
            v16bf bl = load_fragB(brow_lo, d0, g);
            acc0 = __builtin_amdgcn_wmma_f32_16x16x32_bf16(false, ah, false, bh, (short)0, acc0, false, false);
            acc1 = __builtin_amdgcn_wmma_f32_16x16x32_bf16(false, al, false, bh, (short)0, acc1, false, false);
            acc2 = __builtin_amdgcn_wmma_f32_16x16x32_bf16(false, ah, false, bl, (short)0, acc2, false, false);
        }

        const int   code = code0 + ct * 16 + ln16;      // C/D col = lane%16
        const float cs   = csq[code];
#pragma unroll
        for (int v = 0; v < 8; ++v) {
            float s = cs + acc0[v] + acc1[v] + acc2[v]; // = ||c||^2 - 2 x.c
            if (s < bestS[v]) { bestS[v] = s; bestI[v] = code; }  // codes increase -> first-min wins
        }
    }

    // ---- reduce: 16 lanes (same g) share row M = v + 8*g of this row-tile
    __syncthreads();                                    // done reading B; reuse as scratch
    float* redS = (float*)Bhi;                          // [2][MB]
    int*   redI = (int*)Blo;                            // [2][MB]
#pragma unroll
    for (int v = 0; v < 8; ++v) {
        float s = bestS[v]; int bi = bestI[v];
#pragma unroll
        for (int m = 8; m >= 1; m >>= 1) {              // butterfly within 16-lane half
            float os = __shfl_xor(s, m, 32);
            int   oi = __shfl_xor(bi, m, 32);
            if (os < s || (os == s && oi < bi)) { s = os; bi = oi; }
        }
        if (ln16 == 0) {
            const int row = rt * 16 + g * 8 + v;
            redS[ct * MB + row] = s;
            redI[ct * MB + row] = bi;
        }
    }
    __syncthreads();
    if (tid < MB) {
        float s0 = redS[tid];      int i0 = redI[tid];
        float s1 = redS[MB + tid]; int i1 = redI[MB + tid];
        if (s1 < s0 || (s1 == s0 && i1 < i0)) { s0 = s1; i0 = i1; }
        idx_out[rowBase + tid] = i0;
    }
}

// ---------------------------------------------------------------- outputs
__global__ void vq_zero_kernel(float4* __restrict__ p, long long n4) {
    long long i = (long long)blockIdx.x * blockDim.x + threadIdx.x;
    const long long stride = (long long)gridDim.x * blockDim.x;
    const float4 z = make_float4(0.f, 0.f, 0.f, 0.f);
    for (; i < n4; i += stride) p[i] = z;
}

__global__ void vq_scatter_kernel(const int* __restrict__ idx, float* __restrict__ oh) {
    const int n = blockIdx.x * 256 + threadIdx.x;       // 16384 rows
    oh[(long long)n * KCODES + idx[n]] = 1.0f;
}

// quantized (B,C,H,W): out[gid] = codebook[idx[n]][c]; gid contiguous over w -> coalesced
__global__ void vq_quant_kernel(const int* __restrict__ idx,
                                const float* __restrict__ cb,
                                float* __restrict__ out) {
    const long long gid = (long long)blockIdx.x * 256 + threadIdx.x;
    const int w = (int)(gid & 31);
    const int h = (int)((gid >> 5) & 31);
    const int c = (int)((gid >> 10) & 255);
    const long long b = gid >> 18;
    const int n = (int)(b * 1024 + h * 32 + w);
    out[gid] = cb[(long long)idx[n] * DDIM + c];        // gather hits L2 (8MB codebook)
}

extern "C" void kernel_launch(void* const* d_in, const int* in_sizes, int n_in,
                              void* d_out, int out_size, void* d_ws, size_t ws_size,
                              hipStream_t stream) {
    const float* input    = (const float*)d_in[0];      // (16,256,32,32) f32
    const float* codebook = (const float*)d_in[1];      // (8192,256) f32

    float* quant  = (float*)d_out;                      // 16*256*32*32 = 4194304
    float* onehot = (float*)d_out + 4194304;            // 16384*8192

    int*   idx = (int*)d_ws;                            // 16384 ints
    float* csq = (float*)((char*)d_ws + NROWS * sizeof(int)); // 8192 floats

    vq_csq_kernel<<<KCODES / 8, 256, 0, stream>>>(codebook, csq);
    vq_argmin_kernel<<<NROWS / MB, 128, 0, stream>>>(input, codebook, csq, idx);

    const long long oh4 = (long long)NROWS * KCODES / 4;
    vq_zero_kernel<<<4096, 256, 0, stream>>>((float4*)onehot, oh4);
    vq_scatter_kernel<<<NROWS / 256, 256, 0, stream>>>(idx, onehot);
    vq_quant_kernel<<<4194304 / 256, 256, 0, stream>>>(idx, codebook, quant);
}
